// Generator_tailadd_47485158424638
// MI455X (gfx1250) — matrix operand order repaired and checked by reading, not compiled, not run on hardware
//
#include <hip/hip_runtime.h>
#include <cstdint>

typedef _Float16 half_t;
typedef __attribute__((ext_vector_type(16))) _Float16 v16h;
typedef __attribute__((ext_vector_type(8)))  _Float16 v8h;
typedef __attribute__((ext_vector_type(8)))  float    v8f;

#define NH_    8
#define DIM_   384
#define HD_    48
#define NTOK   256
#define NWIN   128
#define MTOT   (NWIN * NTOK)   // 32768
#define HID_   1536
#define QSZ    ((size_t)NWIN * NH_ * NTOK * HD_)   // one Q/K/V region, halves

// ---------------------------------------------------------------------------
// CDNA5 async memory->LDS helpers (ASYNCcnt tracked).
// ---------------------------------------------------------------------------
__device__ __forceinline__ void async_b128(const void* lds_dst,
                                           const void* gbase,
                                           uint32_t byte_off) {
  asm volatile("global_load_async_to_lds_b128 %0, %1, %2 offset:0"
               :
               : "v"((uint32_t)(uintptr_t)lds_dst), "v"(byte_off), "s"(gbase)
               : "memory");
}
__device__ __forceinline__ void wait_async0() {
  asm volatile("s_wait_asynccnt 0" ::: "memory");
}
// Wait until only the newest 4 async ops remain outstanding (per-wave,
// in-order completion -> the previous K-tile's 4 loads have landed).
__device__ __forceinline__ void wait_async4() {
  asm volatile("s_wait_asynccnt 4" ::: "memory");
}

// ---------------------------------------------------------------------------
// WMMA helpers (gfx1250, wave32). 16x16x32 f16 -> f32 accumulate.
// ---------------------------------------------------------------------------
__device__ __forceinline__ v8f wmma16(v16h a, v16h b, v8f c) {
  return __builtin_amdgcn_wmma_f32_16x16x32_f16(false, a, false, b, (short)0, c,
                                                false, false);
}

// A fragment 16x32 f16 from row-major LDS [row][ld]. Unconditional, 2x b128.
__device__ __forceinline__ v16h ldA16(const half_t* base, int row0, int k0,
                                      int ld) {
  int lane = threadIdx.x & 31;
  const half_t* p = base + (size_t)(row0 + (lane & 15)) * ld + k0 +
                    ((lane >> 4) << 3);
  v8h lo = *(const v8h*)p;
  v8h hi = *(const v8h*)(p + 16);
  return __builtin_shufflevector(lo, hi, 0, 1, 2, 3, 4, 5, 6, 7, 8, 9, 10, 11,
                                 12, 13, 14, 15);
}

// B fragment 32x16 from N-major (transposed) LDS: baseT[n][k]. 2x b128.
__device__ __forceinline__ v16h ldBt(const half_t* baseT, int k0, int col0,
                                     int ld) {
  int lane = threadIdx.x & 31;
  const half_t* p = baseT + (size_t)(col0 + (lane & 15)) * ld + k0 +
                    ((lane >> 4) << 3);
  v8h lo = *(const v8h*)p;
  v8h hi = *(const v8h*)(p + 16);
  return __builtin_shufflevector(lo, hi, 0, 1, 2, 3, 4, 5, 6, 7, 8, 9, 10, 11,
                                 12, 13, 14, 15);
}

__device__ __forceinline__ float gelu_f(float x) {
  return 0.5f * x * (1.f + erff(x * 0.70710678118654752f));
}

// ---------------------------------------------------------------------------
// Window gather / scatter (B,H,W,C) <-> (win, tok, C)
// ---------------------------------------------------------------------------
__global__ void gather_kernel(const float* __restrict__ x,
                              float* __restrict__ hbuf) {
  int bi = blockIdx.x;  // w*256 + n
  int w = bi >> 8, n = bi & 255;
  int b = w >> 6, wy = (w >> 3) & 7, wx = w & 7;
  int hh = wy * 16 + (n >> 4), ww = wx * 16 + (n & 15);
  const float4* src =
      (const float4*)(x + (((size_t)b * 128 + hh) * 128 + ww) * DIM_);
  float4* dst = (float4*)(hbuf + (size_t)bi * DIM_);
  dst[threadIdx.x] = src[threadIdx.x];
}

__global__ void scatter_kernel(const float* __restrict__ hbuf,
                               float* __restrict__ out) {
  int bi = blockIdx.x;
  int w = bi >> 8, n = bi & 255;
  int b = w >> 6, wy = (w >> 3) & 7, wx = w & 7;
  int hh = wy * 16 + (n >> 4), ww = wx * 16 + (n & 15);
  const float4* src = (const float4*)(hbuf + (size_t)bi * DIM_);
  float4* dst = (float4*)(out + (((size_t)b * 128 + hh) * 128 + ww) * DIM_);
  dst[threadIdx.x] = src[threadIdx.x];
}

// ---------------------------------------------------------------------------
// Weight convert + transpose: dst[n*K + k] = src[k*N + n] * scale(n)
// (scaleCols>0 folds the attention HD^-0.5 into the Q columns of qkv_w)
// ---------------------------------------------------------------------------
__global__ void cvt_t_kernel(const float* __restrict__ src,
                             half_t* __restrict__ dst, int K, int N,
                             int scaleCols, float scale) {
  int i = blockIdx.x * blockDim.x + threadIdx.x;
  if (i >= K * N) return;
  int k = i / N, n = i - k * N;
  float v = src[i];
  if (n < scaleCols) v *= scale;
  dst[(size_t)n * K + k] = (half_t)v;
}

// ---------------------------------------------------------------------------
// LayerNorm: one wave32 per token (384 = 12 floats/lane), f16 output.
// ---------------------------------------------------------------------------
__global__ void __launch_bounds__(256) ln_kernel(
    const float* __restrict__ hbuf, const float* __restrict__ g,
    const float* __restrict__ bb, half_t* __restrict__ out) {
  int t = threadIdx.x;
  int wave = t >> 5, lane = t & 31;
  size_t row = (size_t)blockIdx.x * 8 + wave;
  const float* xr = hbuf + row * DIM_;
  float v[12];
  float s = 0.f;
#pragma unroll
  for (int i = 0; i < 12; ++i) {
    v[i] = xr[lane + i * 32];
    s += v[i];
  }
#pragma unroll
  for (int m = 16; m > 0; m >>= 1) s += __shfl_xor(s, m, 32);
  float mu = s * (1.f / DIM_);
  float var = 0.f;
#pragma unroll
  for (int i = 0; i < 12; ++i) {
    float d = v[i] - mu;
    var += d * d;
  }
#pragma unroll
  for (int m = 16; m > 0; m >>= 1) var += __shfl_xor(var, m, 32);
  float rstd = rsqrtf(var * (1.f / DIM_) + 1e-5f);
  half_t* o = out + row * DIM_;
#pragma unroll
  for (int i = 0; i < 12; ++i) {
    int c = lane + i * 32;
    o[c] = (half_t)((v[i] - mu) * rstd * g[c] + bb[c]);
  }
}

// ---------------------------------------------------------------------------
// WMMA GEMM: C = A(f16 MxK row-major) @ Bt(f16 NxK row-major, pre-transposed)
// Block tile 128x128, BK=32, 8 waves (4x2), wave tile 32x64 = 2x4 WMMA tiles.
// Double-buffered LDS staging via global_load_async_to_lds_b128:
// next K-tile's 4 async b128/thread are issued before waiting on the current
// tile (s_wait_asynccnt 4 keeps them in flight across the barrier).
// MODE 0: QKV  -> [s][w][h][n][48] f16 (Q,K) ; V stored transposed [w][h][d][n]
// MODE 1: RES  -> hbuf[row*384+col] += acc + bias[col]   (proj / fc2)
// MODE 2: FC1  -> outH[row*N+col] = gelu(acc + bias[col]) f16
// ---------------------------------------------------------------------------
template <int MODE, int N, int K>
__global__ void __launch_bounds__(256) gemm_kernel(
    const half_t* __restrict__ A, const half_t* __restrict__ Bt,
    const float* __restrict__ bias, float* __restrict__ outF,
    half_t* __restrict__ outH) {
  __shared__ __attribute__((aligned(16))) half_t As[2][128 * 40];
  __shared__ __attribute__((aligned(16))) half_t BsT[2][128 * 40];
  const int row0 = blockIdx.y * 128;
  const int col0 = blockIdx.x * 128;
  const int t = threadIdx.x;
  const int wave = __builtin_amdgcn_readfirstlane(t >> 5);
  const int lane = t & 31;
  const int wr = (wave >> 1) * 32;   // 0/32/64/96
  const int wc = (wave & 1) * 64;    // 0/64

  v8f acc[2][4];
  {
    v8f z = {};
#pragma unroll
    for (int i = 0; i < 2; ++i)
#pragma unroll
      for (int j = 0; j < 4; ++j) acc[i][j] = z;
  }

  auto stage = [&](int buf, int k0) {
    int r = t >> 1, seg = (t & 1) * 16;
    {
      uint32_t go = (uint32_t)((((size_t)(row0 + r)) * K + k0 + seg) * 2);
      const half_t* ld = &As[buf][r * 40 + seg];
      async_b128(ld, A, go);
      async_b128(ld + 8, A, go + 16);
    }
    {
      uint32_t go = (uint32_t)((((size_t)(col0 + r)) * K + k0 + seg) * 2);
      const half_t* ld = &BsT[buf][r * 40 + seg];
      async_b128(ld, Bt, go);
      async_b128(ld + 8, Bt, go + 16);
    }
  };

  constexpr int NK = K / 32;
  stage(0, 0);
  for (int ks = 0; ks < NK; ++ks) {
    int cur = ks & 1;
    if (ks + 1 < NK) {
      stage(1 - cur, (ks + 1) * 32);
      wait_async4();
    } else {
      wait_async0();
    }
    __syncthreads();
    {
      const half_t* ab = As[cur];
      const half_t* bb = BsT[cur];
      v16h a0 = ldA16(ab, wr, 0, 40);
      v16h a1 = ldA16(ab, wr + 16, 0, 40);
#pragma unroll
      for (int j = 0; j < 4; ++j) {
        v16h bj = ldBt(bb, 0, wc + j * 16, 40);
        acc[0][j] = wmma16(a0, bj, acc[0][j]);
        acc[1][j] = wmma16(a1, bj, acc[1][j]);
      }
    }
    __syncthreads();
  }

  const int colL = lane & 15;
  const int rbase = (lane >> 4) * 8;
#pragma unroll
  for (int ti = 0; ti < 2; ++ti) {
#pragma unroll
    for (int tj = 0; tj < 4; ++tj) {
      int col = col0 + wc + tj * 16 + colL;
      int rowB = row0 + wr + ti * 16 + rbase;
#pragma unroll
      for (int i = 0; i < 8; ++i) {
        int row = rowB + i;
        float v = acc[ti][tj][i];
        if (MODE == 0) {  // QKV remap (V transposed)
          int w = row >> 8, n = row & 255;
          int s = col / DIM_;
          int r = col - s * DIM_;
          int h = r / HD_;
          int d = r - h * HD_;
          size_t idx;
          if (s == 2)
            idx = 2 * QSZ + (((size_t)w * NH_ + h) * HD_ + d) * NTOK + n;
          else
            idx = (size_t)s * QSZ + (((size_t)w * NH_ + h) * NTOK + n) * HD_ + d;
          outH[idx] = (half_t)v;
        } else if (MODE == 1) {  // residual + bias into f32 stream
          outF[(size_t)row * DIM_ + col] += v + bias[col];
        } else {  // FC1: gelu + bias -> f16
          outH[(size_t)row * N + col] = (half_t)gelu_f(v + bias[col]);
        }
      }
    }
  }
}

// ---------------------------------------------------------------------------
// Fused window attention. One block per (window, head); 256 threads, 8 waves.
// LDS: Q[256][72] f16 (36KB, zero-padded 48..63, reused for V^T),
//      K[256][72] f16 (36KB), S[256][272] f16 (136KB). Total 208KB < 320KB.
// Q already carries the HD^-0.5 scale (folded into the weights).
// ---------------------------------------------------------------------------
__global__ void __launch_bounds__(256) attn_kernel(
    const half_t* __restrict__ qkv, const float* __restrict__ rpb,
    half_t* __restrict__ attnout) {
  __shared__ __attribute__((aligned(16))) half_t Qs[NTOK * 72];
  __shared__ __attribute__((aligned(16))) half_t Ks[NTOK * 72];
  __shared__ __attribute__((aligned(16))) half_t Sb[NTOK * 272];

  const int w = blockIdx.x, h = blockIdx.y;
  const int t = threadIdx.x;
  const int wave = __builtin_amdgcn_readfirstlane(t >> 5);
  const int lane = t & 31;
  const half_t* Qg = qkv + ((size_t)w * NH_ + h) * NTOK * HD_;
  const half_t* Kg = Qg + QSZ;
  const half_t* VgT = Qg + 2 * QSZ;  // [d][n] transposed

  // Stage Q and K rows asynchronously; zero the pad columns 48..63.
  {
    uint32_t go = (uint32_t)((size_t)t * HD_ * 2);
#pragma unroll
    for (int c = 0; c < 3; ++c) {
      async_b128(Qs + t * 72 + c * 8, Qg, go + c * 16);
      async_b128(Ks + t * 72 + c * 8, Kg, go + c * 16);
    }
    v8h z = {};
    *(v8h*)(Qs + t * 72 + 48) = z;
    *(v8h*)(Qs + t * 72 + 56) = z;
    *(v8h*)(Ks + t * 72 + 48) = z;
    *(v8h*)(Ks + t * 72 + 56) = z;
  }
  wait_async0();
  __syncthreads();

  // S = Q @ K^T : 256 16x16 tiles, 32 per wave; K-dim 48 (+16 zero pad).
#pragma unroll
  for (int it = 0; it < 32; ++it) {
    int tid = wave + it * 8;   // wave-uniform, compile-time trip count
    int ti = tid >> 4, tj = tid & 15;
    v8f acc = {};
    acc = wmma16(ldA16(Qs, ti * 16, 0, 72), ldBt(Ks, 0, tj * 16, 72), acc);
    acc = wmma16(ldA16(Qs, ti * 16, 32, 72), ldBt(Ks, 32, tj * 16, 72), acc);
    int colL = lane & 15, rb = (lane >> 4) * 8;
#pragma unroll
    for (int i = 0; i < 8; ++i)
      Sb[(ti * 16 + rb + i) * 272 + tj * 16 + colL] = (half_t)acc[i];
  }
  __syncthreads();

  // Kick off the async V^T copy into the Q region; it overlaps softmax.
  half_t* VsT = Qs;
#pragma unroll
  for (int j = 0; j < 6; ++j) {
    int c = t + 256 * j;          // 8-half chunk id, 1536 total
    int d = c >> 5;
    int n = (c & 31) * 8;
    async_b128(VsT + d * 264 + n, VgT, (uint32_t)(c * 16));
  }

  // Relative-position bias + softmax: one row per thread, 8-half chunks.
  {
    half_t* row = Sb + (size_t)t * 272;
    int i1 = t >> 4, j1 = t & 15;
    float mx = -1e30f;
    for (int c = 0; c < 32; ++c) {
      v8h x = *(const v8h*)(row + c * 8);
#pragma unroll
      for (int j = 0; j < 8; ++j) {
        int m = c * 8 + j;
        int i2 = m >> 4, j2 = m & 15;
        float v = (float)x[j] +
                  rpb[((i1 - i2 + 15) * 31 + (j1 - j2 + 15)) * NH_ + h];
        x[j] = (half_t)v;
        mx = fmaxf(mx, v);
      }
      *(v8h*)(row + c * 8) = x;
    }
    float sum = 0.f;
    for (int c = 0; c < 32; ++c) {
      v8h x = *(const v8h*)(row + c * 8);
#pragma unroll
      for (int j = 0; j < 8; ++j) {
        float e = __expf((float)x[j] - mx);
        x[j] = (half_t)e;
        sum += e;
      }
      *(v8h*)(row + c * 8) = x;
    }
    float inv = 1.f / sum;
    for (int c = 0; c < 32; ++c) {
      v8h x = *(const v8h*)(row + c * 8);
#pragma unroll
      for (int j = 0; j < 8; ++j) x[j] = (half_t)((float)x[j] * inv);
      *(v8h*)(row + c * 8) = x;
    }
  }
  wait_async0();
  __syncthreads();

  // O = P @ V : 48 16x16 tiles (16 rows x 3 dcols), 6 per wave; K over 256.
#pragma unroll
  for (int it = 0; it < 6; ++it) {
    int tid = wave + it * 8;
    int ti = tid / 3, dj = tid - ti * 3;
    v8f acc = {};
#pragma unroll
    for (int k0 = 0; k0 < 256; k0 += 32) {
      acc = wmma16(ldA16(Sb, ti * 16, k0, 272), ldBt(VsT, k0, dj * 16, 264),
                   acc);
    }
    int colL = lane & 15, rb = (lane >> 4) * 8;
#pragma unroll
    for (int i = 0; i < 8; ++i) {
      int n = ti * 16 + rb + i;
      int d = dj * 16 + colL;
      attnout[((size_t)w * NTOK + n) * DIM_ + h * HD_ + d] = (half_t)acc[i];
    }
  }
}

// ---------------------------------------------------------------------------
// Orchestration
// ---------------------------------------------------------------------------
extern "C" void kernel_launch(void* const* d_in, const int* in_sizes, int n_in,
                              void* d_out, int out_size, void* d_ws,
                              size_t ws_size, hipStream_t stream) {
  (void)in_sizes; (void)n_in; (void)out_size; (void)ws_size;
  const float* x      = (const float*)d_in[0];
  const float* qkv_w  = (const float*)d_in[1];
  const float* proj_w = (const float*)d_in[2];
  const float* proj_b = (const float*)d_in[3];
  const float* rpb    = (const float*)d_in[4];
  const float* ln1_g  = (const float*)d_in[5];
  const float* ln1_b  = (const float*)d_in[6];
  const float* ln2_g  = (const float*)d_in[7];
  const float* ln2_b  = (const float*)d_in[8];
  const float* fc1_w  = (const float*)d_in[9];
  const float* fc1_b  = (const float*)d_in[10];
  const float* fc2_w  = (const float*)d_in[11];
  const float* fc2_b  = (const float*)d_in[12];

  char* ws = (char*)d_ws;
  float*  hbuf   = (float*)ws;   ws += (size_t)MTOT * DIM_ * 4;
  half_t* xln16  = (half_t*)ws;  ws += (size_t)MTOT * DIM_ * 2;
  half_t* qkv16  = (half_t*)ws;  ws += 3 * QSZ * 2;
  half_t* attn16 = (half_t*)ws;  ws += (size_t)MTOT * DIM_ * 2;
  half_t* hid16  = (half_t*)ws;  ws += (size_t)MTOT * HID_ * 2;
  half_t* wq16   = (half_t*)ws;  ws += (size_t)2 * DIM_ * 1152 * 2;
  half_t* wp16   = (half_t*)ws;  ws += (size_t)2 * DIM_ * DIM_ * 2;
  half_t* w116   = (half_t*)ws;  ws += (size_t)2 * DIM_ * HID_ * 2;
  half_t* w216   = (half_t*)ws;  ws += (size_t)2 * HID_ * DIM_ * 2;

  const float SCALE = 0.14433756729740645f;  // 48^-0.5

  for (int l = 0; l < 2; ++l) {
    int n;
    n = DIM_ * 1152;
    cvt_t_kernel<<<(n + 255) / 256, 256, 0, stream>>>(
        qkv_w + (size_t)l * n, wq16 + (size_t)l * n, DIM_, 1152, DIM_, SCALE);
    n = DIM_ * DIM_;
    cvt_t_kernel<<<(n + 255) / 256, 256, 0, stream>>>(
        proj_w + (size_t)l * n, wp16 + (size_t)l * n, DIM_, DIM_, 0, 1.f);
    n = DIM_ * HID_;
    cvt_t_kernel<<<(n + 255) / 256, 256, 0, stream>>>(
        fc1_w + (size_t)l * n, w116 + (size_t)l * n, DIM_, HID_, 0, 1.f);
    n = HID_ * DIM_;
    cvt_t_kernel<<<(n + 255) / 256, 256, 0, stream>>>(
        fc2_w + (size_t)l * n, w216 + (size_t)l * n, HID_, DIM_, 0, 1.f);
  }

  gather_kernel<<<MTOT, 96, 0, stream>>>(x, hbuf);

  for (int l = 0; l < 2; ++l) {
    // --- attention sub-block ---
    ln_kernel<<<MTOT / 8, 256, 0, stream>>>(hbuf, ln1_g + l * DIM_,
                                            ln1_b + l * DIM_, xln16);
    gemm_kernel<0, 1152, DIM_><<<dim3(1152 / 128, MTOT / 128), 256, 0, stream>>>(
        xln16, wq16 + (size_t)l * DIM_ * 1152, nullptr, nullptr, qkv16);
    attn_kernel<<<dim3(NWIN, NH_), 256, 0, stream>>>(
        qkv16, rpb + (size_t)l * 961 * NH_, attn16);
    gemm_kernel<1, DIM_, DIM_><<<dim3(DIM_ / 128, MTOT / 128), 256, 0, stream>>>(
        attn16, wp16 + (size_t)l * DIM_ * DIM_, proj_b + l * DIM_, hbuf,
        nullptr);
    // --- MLP sub-block ---
    ln_kernel<<<MTOT / 8, 256, 0, stream>>>(hbuf, ln2_g + l * DIM_,
                                            ln2_b + l * DIM_, xln16);
    gemm_kernel<2, HID_, DIM_><<<dim3(HID_ / 128, MTOT / 128), 256, 0, stream>>>(
        xln16, w116 + (size_t)l * DIM_ * HID_, fc1_b + l * HID_, nullptr,
        hid16);
    gemm_kernel<1, DIM_, HID_><<<dim3(DIM_ / 128, MTOT / 128), 256, 0, stream>>>(
        hid16, w216 + (size_t)l * HID_ * DIM_, fc2_b + l * DIM_, hbuf, nullptr);
  }

  scatter_kernel<<<MTOT, 96, 0, stream>>>(hbuf, (float*)d_out);
}
